// BartDoubleTinyAttention_42898133352595
// MI455X (gfx1250) — matrix operand order repaired and checked by reading, not compiled, not run on hardware
//
#include <hip/hip_runtime.h>

// ---------------------------------------------------------------------------
// CDNA5 (gfx1250) wave32 WMMA implementation of BartDoubleTinyAttention.
// All GEMMs use v_wmma_f32_16x16x32_f16. Attention is flash-style (online
// softmax) so the [B,2048,2048] score matrices never hit memory.
// ---------------------------------------------------------------------------

typedef __attribute__((ext_vector_type(8)))  _Float16 v8h;
typedef __attribute__((ext_vector_type(16))) _Float16 v16h;
typedef __attribute__((ext_vector_type(8)))  float    v8f;

#define DEV __device__ __forceinline__

static DEV v8f wmma_f16f32(v16h a, v16h b, v8f c) {
  // 8 args: (neg_a, A, neg_b, B, c_mod, C, reuse_a, reuse_b)
  return __builtin_amdgcn_wmma_f32_16x16x32_f16(false, a, false, b, (short)0, c,
                                                false, false);
}

// A fragment (16x32 f16), row-major source with row stride `ld`:
//   lane m = lane&15 holds row m; half = lane>>4.
//   element e -> K = (e<8 ? e : e+8) + half*8  => two contiguous 16B loads.
static DEV v16h ld_a(const _Float16* base, int ld, int row0, int k0, int lane) {
  const int half = (lane >> 4) & 1, m = lane & 15;
  const _Float16* p = base + (size_t)(row0 + m) * ld + k0 + half * 8;
  v8h lo = *(const v8h*)p;
  v8h hi = *(const v8h*)(p + 16);
  return __builtin_shufflevector(lo, hi, 0, 1, 2, 3, 4, 5, 6, 7,
                                 8, 9, 10, 11, 12, 13, 14, 15);
}

// B fragment (32x16 f16) where B[k][n] = R[row0+n][k0+k] for row-major R:
//   lane n = lane&15 holds column n; element e -> K = (e<8 ? e : e+8) + half*8
//   => two contiguous 16B loads per lane.
static DEV v16h ld_b(const _Float16* R, int ld, int row0, int k0, int lane) {
  const int half = (lane >> 4) & 1, n = lane & 15;
  const _Float16* p = R + (size_t)(row0 + n) * ld + k0 + half * 8;
  v8h lo = *(const v8h*)p;
  v8h hi = *(const v8h*)(p + 16);
  return __builtin_shufflevector(lo, hi, 0, 1, 2, 3, 4, 5, 6, 7,
                                 8, 9, 10, 11, 12, 13, 14, 15);
}

// ---------------------------------------------------------------------------
// f32 -> f16 elementwise conversion (weights, encoder states)
// ---------------------------------------------------------------------------
__global__ void k_f32_to_f16(const float* __restrict__ s,
                             _Float16* __restrict__ d, int n) {
  int i = blockIdx.x * blockDim.x + threadIdx.x;
  const int stride = gridDim.x * blockDim.x;
  for (; i < n; i += stride) d[i] = (_Float16)s[i];
}

// ---------------------------------------------------------------------------
// Fused LayerNorm(1024) + down-projection to 64: out[16,64] per block.
// 128 threads = 4 waves; LN tile staged in LDS as f16, each wave owns one
// 16-col output tile, K loop = 1024/32 WMMAs.
// ---------------------------------------------------------------------------
__global__ void __launch_bounds__(128)
k_ln_downproj(const float* __restrict__ x, const float* __restrict__ g,
              const float* __restrict__ be, const _Float16* __restrict__ w1h,
              const float* __restrict__ b1, _Float16* __restrict__ out) {
  __shared__ alignas(32) _Float16 As[16 * 1024];
  const int rowBase = blockIdx.x * 16;
  const int t = threadIdx.x;
  {
    const int row = t >> 3, seg = t & 7;  // 8 threads per row
    const float* xr = x + (size_t)(rowBase + row) * 1024 + seg * 128;
    float s = 0.f, ss = 0.f;
    for (int j = 0; j < 128; ++j) { float v = xr[j]; s += v; ss += v * v; }
    for (int msk = 1; msk <= 4; msk <<= 1) {
      s += __shfl_xor(s, msk, 32);
      ss += __shfl_xor(ss, msk, 32);
    }
    const float mean = s * (1.f / 1024.f);
    const float var = ss * (1.f / 1024.f) - mean * mean;
    const float rstd = rsqrtf(var + 1e-5f);
    for (int j = 0; j < 128; ++j) {
      const int c = seg * 128 + j;
      As[row * 1024 + c] = (_Float16)((xr[j] - mean) * rstd * g[c] + be[c]);
    }
  }
  __syncthreads();
  const int lane = t & 31, wave = t >> 5;
  const int nB = wave * 16;
  v8f acc = {0.f, 0.f, 0.f, 0.f, 0.f, 0.f, 0.f, 0.f};
  for (int kk = 0; kk < 32; ++kk) {
    v16h a = ld_a(As, 1024, 0, kk * 32, lane);
    v16h b = ld_b(w1h, 1024, nB, kk * 32, lane);
    acc = wmma_f16f32(a, b, acc);
  }
  const int half = (lane >> 4) & 1, n = lane & 15;
  const int col = nB + n;
  const float bb = b1[col];
  for (int i = 0; i < 8; ++i)
    out[(size_t)(rowBase + half * 8 + i) * 64 + col] = (_Float16)(acc[i] + bb);
}

// ---------------------------------------------------------------------------
// 64x64 projection: out[M,64] = (in[M,64] @ W^T + bias) * scale.
// One wave per 16-row tile. storeT=1 writes output transposed per batch
// ([64][2048], batch = row>>11) so attention P@V B-fragments are contiguous.
// ---------------------------------------------------------------------------
__global__ void __launch_bounds__(32)
k_proj64(const _Float16* __restrict__ in, const _Float16* __restrict__ W,
         const float* __restrict__ bias, _Float16* __restrict__ out,
         float scale, int storeT) {
  const int rowBase = blockIdx.x * 16;
  const int lane = threadIdx.x;
  const int half = (lane >> 4) & 1, n = lane & 15;
  const v16h a0 = ld_a(in, 64, rowBase, 0, lane);
  const v16h a1 = ld_a(in, 64, rowBase, 32, lane);
  for (int j = 0; j < 4; ++j) {
    v8f acc = {0.f, 0.f, 0.f, 0.f, 0.f, 0.f, 0.f, 0.f};
    acc = wmma_f16f32(a0, ld_b(W, 64, j * 16, 0, lane), acc);
    acc = wmma_f16f32(a1, ld_b(W, 64, j * 16, 32, lane), acc);
    const int col = j * 16 + n;
    const float bb = bias[col];
    for (int i = 0; i < 8; ++i) {
      const float v = (acc[i] + bb) * scale;
      const int grow = rowBase + half * 8 + i;
      if (!storeT) {
        out[(size_t)grow * 64 + col] = (_Float16)v;
      } else {
        const int bt = grow >> 11, sl = grow & 2047;
        out[(size_t)bt * (64 * 2048) + (size_t)col * 2048 + sl] = (_Float16)v;
      }
    }
  }
}

// ---------------------------------------------------------------------------
// Flash attention, one wave per 16-row Q tile. Q pre-scaled. K row-major
// [S,64] per batch; V transposed [64,S] per batch. Online softmax: rows of
// the WMMA C layout live in lane-halves, so 16-lane xor reductions give
// per-row max/sum.
// ---------------------------------------------------------------------------
__global__ void __launch_bounds__(32)
k_attn(const _Float16* __restrict__ q, const _Float16* __restrict__ k,
       const _Float16* __restrict__ vt, _Float16* __restrict__ out) {
  __shared__ alignas(32) _Float16 P[16 * 32];
  const int rowBase = blockIdx.x * 16;
  const int batch = rowBase >> 11;  // 2048 rows per batch
  const int lane = threadIdx.x;
  const int half = (lane >> 4) & 1, n = lane & 15;
  const _Float16* kb = k + (size_t)batch * 2048 * 64;
  const _Float16* vtb = vt + (size_t)batch * 64 * 2048;
  const v16h qa0 = ld_a(q, 64, rowBase, 0, lane);
  const v16h qa1 = ld_a(q, 64, rowBase, 32, lane);
  const v8f vz = {0.f, 0.f, 0.f, 0.f, 0.f, 0.f, 0.f, 0.f};
  v8f o0 = vz, o1 = vz, o2 = vz, o3 = vz;
  float m[8], l[8];
  for (int i = 0; i < 8; ++i) { m[i] = -3.0e38f; l[i] = 0.f; }

  for (int s0 = 0; s0 < 2048; s0 += 32) {
    // scores[16,32] = Q(16x64) @ K_chunk^T
    v8f c0 = vz, c1 = vz;
    c0 = wmma_f16f32(qa0, ld_b(kb, 64, s0, 0, lane), c0);
    c0 = wmma_f16f32(qa1, ld_b(kb, 64, s0, 32, lane), c0);
    c1 = wmma_f16f32(qa0, ld_b(kb, 64, s0 + 16, 0, lane), c1);
    c1 = wmma_f16f32(qa1, ld_b(kb, 64, s0 + 16, 32, lane), c1);
    if (s0 + 32 < 2048)  // -> global_prefetch_b8 of next K chunk
      __builtin_prefetch(kb + (size_t)(s0 + 32 + lane) * 64, 0, 1);

    float al[8];
    for (int i = 0; i < 8; ++i) {
      float cm = fmaxf(c0[i], c1[i]);
      for (int msk = 1; msk <= 8; msk <<= 1)
        cm = fmaxf(cm, __shfl_xor(cm, msk, 32));
      const float nm = fmaxf(m[i], cm);
      al[i] = __expf(m[i] - nm);
      m[i] = nm;
    }
    __syncthreads();  // previous iteration's P reads complete
    for (int i = 0; i < 8; ++i) {
      const float p0 = __expf(c0[i] - m[i]);
      const float p1 = __expf(c1[i] - m[i]);
      float rs = p0 + p1;
      for (int msk = 1; msk <= 8; msk <<= 1) rs += __shfl_xor(rs, msk, 32);
      l[i] = l[i] * al[i] + rs;
      o0[i] *= al[i]; o1[i] *= al[i]; o2[i] *= al[i]; o3[i] *= al[i];
      const int r = half * 8 + i;  // C-layout row for this lane/VGPR
      P[r * 32 + n] = (_Float16)p0;
      P[r * 32 + 16 + n] = (_Float16)p1;
    }
    __syncthreads();  // re-layout C -> A fragment through LDS
    const v16h pa = ld_a(P, 32, 0, 0, lane);
    o0 = wmma_f16f32(pa, ld_b(vtb, 2048, 0, s0, lane), o0);
    o1 = wmma_f16f32(pa, ld_b(vtb, 2048, 16, s0, lane), o1);
    o2 = wmma_f16f32(pa, ld_b(vtb, 2048, 32, s0, lane), o2);
    o3 = wmma_f16f32(pa, ld_b(vtb, 2048, 48, s0, lane), o3);
  }

  for (int i = 0; i < 8; ++i) {
    const float inv = 1.f / l[i];
    const size_t rb = (size_t)(rowBase + half * 8 + i) * 64;
    out[rb + 0 + n]  = (_Float16)(o0[i] * inv);
    out[rb + 16 + n] = (_Float16)(o1[i] * inv);
    out[rb + 32 + n] = (_Float16)(o2[i] * inv);
    out[rb + 48 + n] = (_Float16)(o3[i] * inv);
  }
}

// ---------------------------------------------------------------------------
// Up-projection + scaled residual: out = x + 0.001*(h @ w2^T + b2).
// Grid (512 row tiles, 16 col tiles of 64). One wave per tile.
// ---------------------------------------------------------------------------
__global__ void __launch_bounds__(32)
k_upproj(const _Float16* __restrict__ h, const _Float16* __restrict__ w2h,
         const float* __restrict__ b2, const float* __restrict__ x,
         float* __restrict__ out) {
  const int rowBase = blockIdx.x * 16;
  const int colBase = blockIdx.y * 64;
  const int lane = threadIdx.x;
  const int half = (lane >> 4) & 1, n = lane & 15;
  const v16h a0 = ld_a(h, 64, rowBase, 0, lane);
  const v16h a1 = ld_a(h, 64, rowBase, 32, lane);
  for (int j = 0; j < 4; ++j) {
    v8f acc = {0.f, 0.f, 0.f, 0.f, 0.f, 0.f, 0.f, 0.f};
    acc = wmma_f16f32(a0, ld_b(w2h, 64, colBase + j * 16, 0, lane), acc);
    acc = wmma_f16f32(a1, ld_b(w2h, 64, colBase + j * 16, 32, lane), acc);
    const int col = colBase + j * 16 + n;
    const float bb = b2[col];
    for (int i = 0; i < 8; ++i) {
      const size_t idx = (size_t)(rowBase + half * 8 + i) * 1024 + col;
      out[idx] = x[idx] + 0.001f * (acc[i] + bb);
    }
  }
}

// ---------------------------------------------------------------------------
extern "C" void kernel_launch(void* const* d_in, const int* in_sizes, int n_in,
                              void* d_out, int out_size, void* d_ws,
                              size_t ws_size, hipStream_t stream) {
  (void)in_sizes; (void)n_in; (void)out_size; (void)ws_size;
  const float* x   = (const float*)d_in[0];
  const float* enc = (const float*)d_in[1];
  const float* lng = (const float*)d_in[2];
  const float* lnb = (const float*)d_in[3];
  const float* w1  = (const float*)d_in[4];
  const float* b1  = (const float*)d_in[5];
  const float* wq1 = (const float*)d_in[6];  const float* bq1 = (const float*)d_in[7];
  const float* wk1 = (const float*)d_in[8];  const float* bk1 = (const float*)d_in[9];
  const float* wv1 = (const float*)d_in[10]; const float* bv1 = (const float*)d_in[11];
  const float* wo1 = (const float*)d_in[12]; const float* bo1 = (const float*)d_in[13];
  const float* wq2 = (const float*)d_in[14]; const float* bq2 = (const float*)d_in[15];
  const float* wk2 = (const float*)d_in[16]; const float* bk2 = (const float*)d_in[17];
  const float* wv2 = (const float*)d_in[18]; const float* bv2 = (const float*)d_in[19];
  const float* wo2 = (const float*)d_in[20]; const float* bo2 = (const float*)d_in[21];
  const float* w2  = (const float*)d_in[22]; const float* b2  = (const float*)d_in[23];
  float* out = (float*)d_out;

  char* ws = (char*)d_ws;
  size_t off = 0;
  auto take = [&](size_t bytes) -> _Float16* {
    char* p = ws + off;
    off += (bytes + 255) & ~(size_t)255;
    return (_Float16*)p;
  };
  const size_t ACT = (size_t)8192 * 64 * sizeof(_Float16);  // 1 MB
  _Float16* enc16 = take((size_t)8192 * 64 * 2);
  _Float16* w1h   = take((size_t)64 * 1024 * 2);
  _Float16* w2h   = take((size_t)1024 * 64 * 2);
  _Float16* wq1h = take(4096 * 2); _Float16* wk1h = take(4096 * 2);
  _Float16* wv1h = take(4096 * 2); _Float16* wo1h = take(4096 * 2);
  _Float16* wq2h = take(4096 * 2); _Float16* wk2h = take(4096 * 2);
  _Float16* wv2h = take(4096 * 2); _Float16* wo2h = take(4096 * 2);
  _Float16* hA  = take(ACT);   // down-proj output / final attn-space act
  _Float16* hB  = take(ACT);   // cross-attn block output
  _Float16* qB  = take(ACT);
  _Float16* kB  = take(ACT);
  _Float16* vtB = take(ACT);   // V transposed per batch [64][2048]
  _Float16* aB  = take(ACT);   // attention output (pre-Wo)

  const float SCALE = 0.125f;  // 64^-0.5

#define CONV(src, dst, nelem)                                                  \
  k_f32_to_f16<<<dim3(((nelem) + 255) / 256), dim3(256), 0, stream>>>(src, dst, nelem)
  CONV(enc, enc16, 8192 * 64);
  CONV(w1, w1h, 64 * 1024);
  CONV(w2, w2h, 1024 * 64);
  CONV(wq1, wq1h, 4096); CONV(wk1, wk1h, 4096);
  CONV(wv1, wv1h, 4096); CONV(wo1, wo1h, 4096);
  CONV(wq2, wq2h, 4096); CONV(wk2, wk2h, 4096);
  CONV(wv2, wv2h, 4096); CONV(wo2, wo2h, 4096);
#undef CONV

  // LayerNorm + down-proj: h = LN(x) @ w1^T + b1
  k_ln_downproj<<<dim3(512), dim3(128), 0, stream>>>(x, lng, lnb, w1h, b1, hA);

  // Cross-attention (q from hA, k/v from encoder states)
  k_proj64<<<dim3(512), dim3(32), 0, stream>>>(hA, wq1h, bq1, qB, SCALE, 0);
  k_proj64<<<dim3(512), dim3(32), 0, stream>>>(enc16, wk1h, bk1, kB, 1.f, 0);
  k_proj64<<<dim3(512), dim3(32), 0, stream>>>(enc16, wv1h, bv1, vtB, 1.f, 1);
  k_attn<<<dim3(512), dim3(32), 0, stream>>>(qB, kB, vtB, aB);
  k_proj64<<<dim3(512), dim3(32), 0, stream>>>(aB, wo1h, bo1, hB, 1.f, 0);

  // Self-attention (q/k/v all from hB)
  k_proj64<<<dim3(512), dim3(32), 0, stream>>>(hB, wq2h, bq2, qB, SCALE, 0);
  k_proj64<<<dim3(512), dim3(32), 0, stream>>>(hB, wk2h, bk2, kB, 1.f, 0);
  k_proj64<<<dim3(512), dim3(32), 0, stream>>>(hB, wv2h, bv2, vtB, 1.f, 1);
  k_attn<<<dim3(512), dim3(32), 0, stream>>>(qB, kB, vtB, aB);
  k_proj64<<<dim3(512), dim3(32), 0, stream>>>(aB, wo2h, bo2, hA, 1.f, 0);

  // Up-proj + scaled residual
  k_upproj<<<dim3(512, 16), dim3(32), 0, stream>>>(hA, w2h, b2, x, out);
}
